// Attention2d_24627342475445
// MI455X (gfx1250) — matrix operand order repaired and checked by reading, not compiled, not run on hardware
//
#include <hip/hip_runtime.h>
#include <hip/hip_bf16.h>

// Attention2d for MI455X (gfx1250, wave32, WMMA f32_16x16x32_f16).
// GroupNorm -> QKV GEMM (WMMA) -> flash attention (WMMA) -> proj GEMM (WMMA) + residual.
// Token-major activations => all LDS staging is contiguous async-to-LDS B128 (ASYNCcnt),
// double-buffered against WMMA compute. Softmax row-sums computed on the matrix engine.

#define HW      4096
#define C_CH    256
#define NHEADS  4
#define HDIM    64
#define BATCH   2

typedef __attribute__((ext_vector_type(4)))  int      v4i;
typedef __attribute__((ext_vector_type(8)))  _Float16 v8h;
typedef __attribute__((ext_vector_type(16))) _Float16 v16h;
typedef __attribute__((ext_vector_type(8)))  float    v8f;

#ifdef __has_builtin
#if __has_builtin(__builtin_amdgcn_global_load_async_to_lds_b128) && \
    __has_builtin(__builtin_amdgcn_s_wait_asynccnt)
#define HAVE_ASYNC_LDS 1
#endif
#if __has_builtin(__builtin_amdgcn_permlane16)
#define HAVE_PERMLANE16 1
#endif
#endif
#ifndef HAVE_ASYNC_LDS
#define HAVE_ASYNC_LDS 0
#endif
#ifndef HAVE_PERMLANE16
#define HAVE_PERMLANE16 0
#endif

#define GLOBAL_AS __attribute__((address_space(1)))
#define LDS_AS    __attribute__((address_space(3)))

// 16-byte global->LDS copy; async (ASYNCcnt) when toolchain exposes it.
__device__ __forceinline__ void copy16_g2l(const _Float16* g, _Float16* l) {
#if HAVE_ASYNC_LDS
  __builtin_amdgcn_global_load_async_to_lds_b128((GLOBAL_AS v4i*)g, (LDS_AS v4i*)l, 0, 0);
#else
  *(v8h*)l = *(const v8h*)g;
#endif
}
__device__ __forceinline__ void copy_wait() {
#if HAVE_ASYNC_LDS
  __builtin_amdgcn_s_wait_asynccnt(0);
#endif
}

#if HAVE_PERMLANE16
__device__ __forceinline__ float pl16(float v, unsigned s0, unsigned s1) {
  unsigned u = __float_as_uint(v);
  unsigned r = __builtin_amdgcn_permlane16(u, u, s0, s1, false, false);
  return __uint_as_float(r);
}
#endif

// Max over each 16-lane group (both wave halves independently).
__device__ __forceinline__ float rowmax16(float v) {
#if HAVE_PERMLANE16
  v = fmaxf(v, pl16(v, 0x67452301u, 0xEFCDAB89u));  // xor 1
  v = fmaxf(v, pl16(v, 0x54761032u, 0xDCFE98BAu));  // xor 2
  v = fmaxf(v, pl16(v, 0x32107654u, 0xBA98FEDCu));  // xor 4
  v = fmaxf(v, pl16(v, 0xFEDCBA98u, 0x76543210u));  // xor 8
#else
  v = fmaxf(v, __shfl_xor(v, 1, 16));
  v = fmaxf(v, __shfl_xor(v, 2, 16));
  v = fmaxf(v, __shfl_xor(v, 4, 16));
  v = fmaxf(v, __shfl_xor(v, 8, 16));
#endif
  return v;
}

__device__ __forceinline__ v8f zero8() {
  v8f z = {0.f, 0.f, 0.f, 0.f, 0.f, 0.f, 0.f, 0.f};
  return z;
}

// A fragment (16x32, MxK), tile row-major [M][K], leading dim `ld` halves.
__device__ __forceinline__ v16h load_frag_a(const _Float16* base, int ld, int k0, int lane) {
  const _Float16* p = base + (lane & 15) * ld + k0 + ((lane >> 4) << 3);
  v8h lo = *(const v8h*)p;
  v8h hi = *(const v8h*)(p + 16);
  v16h r;
#pragma unroll
  for (int i = 0; i < 8; ++i) { r[i] = lo[i]; r[i + 8] = hi[i]; }
  return r;
}

// B fragment (32x16, KxN), tile stored TRANSPOSED [N][K], leading dim `ld` halves.
__device__ __forceinline__ v16h load_frag_b(const _Float16* base, size_t ld, int k0, int lane) {
  const _Float16* p = base + (size_t)(lane & 15) * ld + k0 + ((lane >> 4) << 3);
  v8h lo = *(const v8h*)p;
  v8h hi = *(const v8h*)(p + 16);
  v16h r;
#pragma unroll
  for (int i = 0; i < 8; ++i) { r[i] = lo[i]; r[i + 8] = hi[i]; }
  return r;
}

__device__ __forceinline__ v8f wmma_f16(v16h a, v16h b, v8f c) {
  return __builtin_amdgcn_wmma_f32_16x16x32_f16(false, a, false, b, (short)0, c, false, false);
}

// ---------------------------------------------------------------- convert
__global__ __launch_bounds__(256) void cvt_f32_f16(const float* __restrict__ in,
                                                   _Float16* __restrict__ out, int n) {
  int i = blockIdx.x * 256 + threadIdx.x;
  if (i < n) out[i] = (_Float16)in[i];
}

// ---------------------------------------------------------------- GroupNorm
// One block per (batch, group). Writes token-major xnT[b][p][C] (f16).
__global__ __launch_bounds__(256) void groupnorm_kernel(const float* __restrict__ x,
                                                        const float* __restrict__ w,
                                                        const float* __restrict__ bgn,
                                                        _Float16* __restrict__ xnT) {
  const int t = threadIdx.x;
  const int b = blockIdx.x >> 5;
  const int g = blockIdx.x & 31;
  const int CPG = C_CH / 32;  // 8
  const size_t base = ((size_t)b * C_CH + (size_t)g * CPG) * HW;
  const int N = CPG * HW;  // 32768
  float s = 0.f, q = 0.f;
  for (int i = t; i < N; i += 256) {
    float v = x[base + i];
    s += v;
    q += v * v;
  }
  __shared__ float rs[256], rq[256];
  rs[t] = s;
  rq[t] = q;
  __syncthreads();
  for (int st = 128; st > 0; st >>= 1) {
    if (t < st) { rs[t] += rs[t + st]; rq[t] += rq[t + st]; }
    __syncthreads();
  }
  const float mu = rs[0] / (float)N;
  const float var = rq[0] / (float)N - mu * mu;
  const float rinv = rsqrtf(var + 1e-5f);
  for (int i = t; i < N; i += 256) {
    int c = g * CPG + (i >> 12);
    int p = i & (HW - 1);
    float v = (x[base + i] - mu) * rinv * w[c] + bgn[c];
    xnT[((size_t)b * HW + p) * C_CH + c] = (_Float16)v;
  }
}

// ---------------------------------------------------------------- WMMA GEMM
// acc[M][N] = W[M][K] * XT[b][N][K]^T + bias[M]. Double-buffered async LDS staging.
// MODE 0 (QKV): rows<512 (Q,K) -> packed v8h into qkvT[b][p][512], Q scaled 1/8;
//               rows>=512 (V) -> v_h[b][d][p] channel-major.
// MODE 1 (proj): f32 out[b][c][p] = acc + bias + resid.
template <int MODE>
__global__ __launch_bounds__(256) void gemm_kernel(const _Float16* __restrict__ W,
                                                   const _Float16* __restrict__ XT,
                                                   const float* __restrict__ bias,
                                                   const float* __restrict__ resid,
                                                   _Float16* __restrict__ outA,
                                                   _Float16* __restrict__ outB,
                                                   float* __restrict__ out_f,
                                                   int M, int K) {
  const int t = threadIdx.x;
  const int n0 = blockIdx.x * 128;
  const int m0 = blockIdx.y * 128;
  const int b = blockIdx.z;
  const _Float16* Xb = XT + (size_t)b * HW * K;  // [p][K]

  __shared__ _Float16 As[2][128 * 32];  // [M][K]
  __shared__ _Float16 Bs[2][128 * 32];  // [N][K]

  const int waveId = t >> 5, lane = t & 31;
  const int wm = waveId >> 2, wn = waveId & 3;

  auto issue_tiles = [&](int k0, int buf) {
    int row = t >> 1;
    int col = (t & 1) << 4;
    const _Float16* src = W + (size_t)(m0 + row) * K + k0 + col;
    copy16_g2l(src, &As[buf][row * 32 + col]);
    copy16_g2l(src + 8, &As[buf][row * 32 + col + 8]);
#pragma unroll
    for (int it = 0; it < 2; ++it) {
      int u = t + it * 256;
      int nr = u >> 2;        // 0..127
      int kc = (u & 3) << 3;  // 0,8,16,24
      copy16_g2l(Xb + (size_t)(n0 + nr) * K + k0 + kc, &Bs[buf][nr * 32 + kc]);
    }
  };

  v8f acc[4][2];
#pragma unroll
  for (int i = 0; i < 4; ++i)
#pragma unroll
    for (int j = 0; j < 2; ++j) acc[i][j] = zero8();

  issue_tiles(0, 0);
  copy_wait();
  __syncthreads();

  for (int k0 = 0; k0 < K; k0 += 32) {
    const int cur = (k0 >> 5) & 1;
    if (k0 + 32 < K) issue_tiles(k0 + 32, cur ^ 1);  // overlap DMA with WMMA
    if (k0 + 64 < K) {
      __builtin_prefetch((const void*)(W + (size_t)(m0 + (t >> 1)) * K + k0 + 64), 0, 1);
      __builtin_prefetch((const void*)(Xb + (size_t)(n0 + (t >> 1)) * K + k0 + 64), 0, 1);
    }
    v16h bfr[2];
#pragma unroll
    for (int j = 0; j < 2; ++j)
      bfr[j] = load_frag_b(&Bs[cur][(wn * 32 + j * 16) * 32], 32, 0, lane);
#pragma unroll
    for (int i = 0; i < 4; ++i) {
      v16h a = load_frag_a(&As[cur][(wm * 64 + i * 16) * 32], 32, 0, lane);
#pragma unroll
      for (int j = 0; j < 2; ++j) acc[i][j] = wmma_f16(a, bfr[j], acc[i][j]);
    }
    copy_wait();
    __syncthreads();
  }

  const int rb = (lane >> 4) << 3;  // C/D layout: row = vr + 8*(lane>=16)
#pragma unroll
  for (int i = 0; i < 4; ++i)
#pragma unroll
    for (int j = 0; j < 2; ++j) {
      const int colp = n0 + wn * 32 + j * 16 + (lane & 15);  // token p
      const int row0 = m0 + wm * 64 + i * 16 + rb;           // first of 8 channels
      if (MODE == 0) {
        if (m0 < 2 * C_CH) {  // Q or K rows -> token-major packed store
          const float sc = (m0 < C_CH) ? 0.125f : 1.0f;      // fold 1/sqrt(64) into Q
          v8h pack;
#pragma unroll
          for (int vr = 0; vr < 8; ++vr)
            pack[vr] = (_Float16)((acc[i][j][vr] + bias[row0 + vr]) * sc);
          *(v8h*)(outA + ((size_t)b * HW + colp) * (2 * C_CH) + row0) = pack;
        } else {              // V rows -> channel-major [d][p]
#pragma unroll
          for (int vr = 0; vr < 8; ++vr) {
            float v = acc[i][j][vr] + bias[row0 + vr];
            outB[((size_t)b * C_CH + (row0 + vr - 2 * C_CH)) * HW + colp] = (_Float16)v;
          }
        }
      } else {
#pragma unroll
        for (int vr = 0; vr < 8; ++vr) {
          size_t off = ((size_t)b * C_CH + row0 + vr) * HW + colp;
          out_f[off] = acc[i][j][vr] + bias[row0 + vr] + resid[off];
        }
      }
    }
}

// ---------------------------------------------------------------- Flash attention
// Per (b,h): S = Q K^T (Q pre-scaled), P = online softmax, O += P V^T.
// 128 query rows per block, 8 waves each owning 16 full rows.
// Row-sum l computed by an extra WMMA against an all-ones B matrix (every column of the
// result equals the row sum, already in C-layout row indexing) -> no sum shuffles.
__global__ __launch_bounds__(256) void flash_kernel(const _Float16* __restrict__ qkvT,
                                                    const _Float16* __restrict__ v_h,
                                                    _Float16* __restrict__ attnT) {
  const int t = threadIdx.x;
  const int i0 = blockIdx.x * 128;
  const int h = blockIdx.y;
  const int b = blockIdx.z;
  const _Float16* QT = qkvT + (size_t)b * HW * (2 * C_CH) + h * HDIM;         // [p][512]
  const _Float16* KT = qkvT + (size_t)b * HW * (2 * C_CH) + C_CH + h * HDIM;  // K cols at 256+
  const _Float16* Vg = v_h + ((size_t)b * C_CH + h * HDIM) * HW;              // [d][p]

  __shared__ _Float16 Qs[128 * 64];     // [i][d]
  __shared__ _Float16 Ks[2][64 * 64];   // [j][d], double-buffered
  __shared__ _Float16 Ps[8 * 16 * 64];  // per-wave P tile [i][j]

  const int waveId = t >> 5, lane = t & 31;
  const int rb = (lane >> 4) << 3;

  auto issue_k = [&](int j0, int buf) {
#pragma unroll
    for (int it = 0; it < 2; ++it) {
      int u = t + it * 256;   // 512 chunks
      int jr = u >> 3;        // 0..63
      int dc = (u & 7) << 3;  // 0..56
      copy16_g2l(KT + (size_t)(j0 + jr) * (2 * C_CH) + dc, &Ks[buf][jr * 64 + dc]);
    }
  };

  // Q block (already scaled by 1/8) + first K tile.
#pragma unroll
  for (int it = 0; it < 4; ++it) {
    int u = t + it * 256;   // 1024 chunks
    int ir = u >> 3;        // 0..127
    int dc = (u & 7) << 3;  // 0..56
    copy16_g2l(QT + (size_t)(i0 + ir) * (2 * C_CH) + dc, Qs + ir * 64 + dc);
  }
  issue_k(0, 0);
  copy_wait();
  __syncthreads();

  float m_r[8];
#pragma unroll
  for (int vr = 0; vr < 8; ++vr) m_r[vr] = -1e30f;
  v8f o[4];
#pragma unroll
  for (int dt = 0; dt < 4; ++dt) o[dt] = zero8();
  v8f o_l = zero8();  // running row-sum accumulator (via ones-WMMA)

  v16h ones;
#pragma unroll
  for (int i = 0; i < 16; ++i) ones[i] = (_Float16)1.0f;

  const _Float16* Qw = Qs + waveId * 16 * 64;
  _Float16* Pw = Ps + waveId * 16 * 64;  // wave-private

  for (int j0 = 0; j0 < HW; j0 += 64) {
    const int cur = (j0 >> 6) & 1;
    if (j0 + 64 < HW) issue_k(j0 + 64, cur ^ 1);  // overlap DMA with compute
    if (j0 + 128 < HW)
      __builtin_prefetch((const void*)(KT + (size_t)(j0 + 128 + (t >> 2)) * (2 * C_CH) +
                                       ((t & 3) << 4)), 0, 1);

    // S tile 16x64 per wave (4 n-tiles, K=64 in two steps).
    v8f s[4];
#pragma unroll
    for (int nt = 0; nt < 4; ++nt) s[nt] = zero8();
#pragma unroll
    for (int kk = 0; kk < 64; kk += 32) {
      v16h a = load_frag_a(Qw, 64, kk, lane);
#pragma unroll
      for (int nt = 0; nt < 4; ++nt) {
        v16h bb = load_frag_b(&Ks[cur][nt * 16 * 64], 64, kk, lane);
        s[nt] = wmma_f16(a, bb, s[nt]);
      }
    }

    // Online softmax: only the row-MAX needs a cross-lane reduction (pure VALU).
#pragma unroll
    for (int vr = 0; vr < 8; ++vr) {
      float mx = fmaxf(fmaxf(s[0][vr], s[1][vr]), fmaxf(s[2][vr], s[3][vr]));
      mx = rowmax16(mx);
      float mnew = fmaxf(m_r[vr], mx);
      float sc = __expf(m_r[vr] - mnew);
      m_r[vr] = mnew;
#pragma unroll
      for (int nt = 0; nt < 4; ++nt) s[nt][vr] = __expf(s[nt][vr] - mnew);
#pragma unroll
      for (int dt = 0; dt < 4; ++dt) o[dt][vr] *= sc;
      o_l[vr] *= sc;
    }

    // P (C-layout f32) -> wave-private LDS [i][j] f16 (A-fragment source).
#pragma unroll
    for (int vr = 0; vr < 8; ++vr)
#pragma unroll
      for (int nt = 0; nt < 4; ++nt)
        Pw[(vr + rb) * 64 + nt * 16 + (lane & 15)] = (_Float16)s[nt][vr];

    // O += P * V^T and l += P * 1 on the matrix engine.
#pragma unroll
    for (int kk = 0; kk < 64; kk += 32) {
      v16h a = load_frag_a(Pw, 64, kk, lane);
      o_l = wmma_f16(a, ones, o_l);
#pragma unroll
      for (int dt = 0; dt < 4; ++dt) {
        v16h bv = load_frag_b(Vg + (size_t)dt * 16 * HW, HW, j0 + kk, lane);
        o[dt] = wmma_f16(a, bv, o[dt]);
      }
    }
    copy_wait();
    __syncthreads();
  }

  // Normalize, store token-major attnT[b][p][256] (feeds proj B-tiles).
#pragma unroll
  for (int dt = 0; dt < 4; ++dt)
#pragma unroll
    for (int vr = 0; vr < 8; ++vr) {
      float inv = 1.0f / o_l[vr];
      int p = i0 + waveId * 16 + vr + rb;
      int c = h * HDIM + dt * 16 + (lane & 15);
      attnT[((size_t)b * HW + p) * C_CH + c] = (_Float16)(o[dt][vr] * inv);
    }
}

// ---------------------------------------------------------------- launch
extern "C" void kernel_launch(void* const* d_in, const int* in_sizes, int n_in,
                              void* d_out, int out_size, void* d_ws, size_t ws_size,
                              hipStream_t stream) {
  const float* x     = (const float*)d_in[0];
  const float* gnw   = (const float*)d_in[1];
  const float* gnb   = (const float*)d_in[2];
  const float* qkvw  = (const float*)d_in[3];  // [768][256]
  const float* qkvb  = (const float*)d_in[4];  // [768]
  const float* projw = (const float*)d_in[5];  // [256][256]
  const float* projb = (const float*)d_in[6];  // [256]
  float* out = (float*)d_out;

  // Workspace carve (f16): ~20.5 MiB.
  _Float16* ws     = (_Float16*)d_ws;
  _Float16* xnT    = ws;                                      // B*HW*C   (token-major)
  _Float16* qkvT   = xnT + (size_t)BATCH * HW * C_CH;         // B*HW*512 (Q|K token-major)
  _Float16* v_h    = qkvT + (size_t)BATCH * HW * 2 * C_CH;    // B*C*HW   (V channel-major)
  _Float16* attnT  = v_h + (size_t)BATCH * C_CH * HW;         // B*HW*C   (token-major)
  _Float16* wq_h   = attnT + (size_t)BATCH * HW * C_CH;       // 768*256
  _Float16* wp_h   = wq_h + (size_t)3 * C_CH * C_CH;          // 256*256

  cvt_f32_f16<<<(3 * C_CH * C_CH + 255) / 256, 256, 0, stream>>>(qkvw, wq_h, 3 * C_CH * C_CH);
  cvt_f32_f16<<<(C_CH * C_CH + 255) / 256, 256, 0, stream>>>(projw, wp_h, C_CH * C_CH);

  groupnorm_kernel<<<BATCH * 32, 256, 0, stream>>>(x, gnw, gnb, xnT);

  gemm_kernel<0><<<dim3(HW / 128, (3 * C_CH) / 128, BATCH), 256, 0, stream>>>(
      wq_h, xnT, qkvb, nullptr, qkvT, v_h, nullptr, 3 * C_CH, C_CH);

  flash_kernel<<<dim3(HW / 128, NHEADS, BATCH), 256, 0, stream>>>(qkvT, v_h, attnT);

  gemm_kernel<1><<<dim3(HW / 128, C_CH / 128, BATCH), 256, 0, stream>>>(
      wp_h, attnT, projb, x, nullptr, nullptr, out, C_CH, C_CH);
}
